// APN_68650757259606
// MI455X (gfx1250) — matrix operand order repaired and checked by reading, not compiled, not run on hardware
//
#include <hip/hip_runtime.h>
#include <hip/hip_bf16.h>
#include <cstdint>
#include <cstddef>

// ---------------------------------------------------------------------------
// Types for CDNA5 WMMA (wave32, v_wmma_f32_16x16x32_bf16)
// ---------------------------------------------------------------------------
typedef __attribute__((ext_vector_type(16))) __bf16 v16bf;
typedef __attribute__((ext_vector_type(8)))  __bf16 v8bf;
typedef __attribute__((ext_vector_type(8)))  float  v8f;

typedef unsigned short ushort_t;
typedef unsigned int   uint_t;

union FragU { v16bf v; v8bf h[2]; };

#define NB      32768      // nodes in batched graph (B * N_OBJ)
#define NBATCH  256
#define NOBJ    128
#define HID     256
#define NEDGES  262144
#define NETYPES 4

// ---------------------------------------------------------------------------
// bf16 helpers (manual round-to-nearest-even; storage as ushort)
// ---------------------------------------------------------------------------
__device__ __forceinline__ ushort_t f2b(float f) {
  uint_t u = __float_as_uint(f);
  uint_t r = (u + 0x7FFFu + ((u >> 16) & 1u)) >> 16;
  return (ushort_t)r;
}
__device__ __forceinline__ float b2f(ushort_t h) {
  return __uint_as_float(((uint_t)h) << 16);
}

// ---------------------------------------------------------------------------
// A-fragment loader: row-major bf16 [M, strideA]; 16x32 bf16 tile.
// ISA layout (16-bit A 16x32): lanes 0-15 -> rows, K 0..7 & 16..23;
// lanes 16-31 -> rows, K 8..15 & 24..31.
// ---------------------------------------------------------------------------
__device__ __forceinline__ v16bf load_afrag(const ushort_t* __restrict__ A,
                                            int row0, int M, int strideA,
                                            int k0, int lane) {
  int r = row0 + (lane & 15);
  if (r >= M) r = M - 1;              // clamp: garbage rows never stored
  const ushort_t* p = A + (size_t)r * strideA + k0 + ((lane >> 4) << 3);
  FragU u;
  u.h[0] = *(const v8bf*)(p);
  u.h[1] = *(const v8bf*)(p + 16);
  return u.v;
}

// B-fragment: packed buffer, tile (kt,nt): lane L owns 16 contiguous bf16.
__device__ __forceinline__ v16bf load_bfrag(const ushort_t* __restrict__ Bp,
                                            int kt, int nt, int NT, int lane) {
  const ushort_t* p = Bp + (((size_t)kt * NT + nt) * 32 + lane) * 16;
  FragU u;
  u.h[0] = *(const v8bf*)(p);
  u.h[1] = *(const v8bf*)(p + 8);
  return u.v;
}

// ---------------------------------------------------------------------------
// Generic WMMA GEMM: C[M,Np] = act(A[M,Kp]_bf16 @ Bp + bias)
// ACT: 0 none, 1 relu, 2 prelu(*alphap), 3 sigmoid
// block = 256 threads = 8 waves; each wave computes NTW consecutive 16x16
// N-tiles from one shared A fragment (A traffic cut NTW x, NTW independent
// accumulation chains hide WMMA latency).
// grid = (ceil(M/128), Np/(16*NTW))
// ---------------------------------------------------------------------------
template<int ACT, int NTW>
__global__ void k_gemm(const ushort_t* __restrict__ A, int strideA,
                       const ushort_t* __restrict__ Bp,
                       const float* __restrict__ bias,
                       float* __restrict__ outF, ushort_t* __restrict__ outB,
                       int M, int Np, int Kp,
                       const float* __restrict__ alphap) {
  const int lane = threadIdx.x & 31;
  const int wave = threadIdx.x >> 5;
  const int row0 = (blockIdx.x * 8 + wave) * 16;
  const int nt0  = blockIdx.y * NTW;
  const int NT   = Np >> 4;
  const int KT   = Kp >> 5;
  v8f c[NTW];
#pragma unroll
  for (int j = 0; j < NTW; ++j) c[j] = v8f{0.f,0.f,0.f,0.f,0.f,0.f,0.f,0.f};
  for (int kt = 0; kt < KT; ++kt) {
    v16bf a = load_afrag(A, row0, M, strideA, kt << 5, lane);
#pragma unroll
    for (int j = 0; j < NTW; ++j) {
      v16bf b = load_bfrag(Bp, kt, nt0 + j, NT, lane);
      c[j] = __builtin_amdgcn_wmma_f32_16x16x32_bf16(false, a, false, b,
                                                     (short)0, c[j], false, false);
    }
  }
  float alpha = 0.f;
  if (ACT == 2) alpha = *alphap;
  const int mb = row0 + ((lane >> 4) << 3);
#pragma unroll
  for (int j = 0; j < NTW; ++j) {
    const int n  = ((nt0 + j) << 4) + (lane & 15);
    const float bv = bias ? bias[n] : 0.f;
#pragma unroll
    for (int i = 0; i < 8; ++i) {
      int m = mb + i;
      if (m < M) {
        float v = c[j][i] + bv;
        if (ACT == 1) v = fmaxf(v, 0.f);
        if (ACT == 2) v = (v >= 0.f) ? v : alpha * v;
        if (ACT == 3) v = 1.f / (1.f + __expf(-v));
        if (outF) outF[(size_t)m * Np + n] = v;
        if (outB) outB[(size_t)m * Np + n] = f2b(v);
      }
    }
  }
}

// Gated GEMM: out = sigmoid(A@G) * (A@W)   (per-node RGCN message)
// NTW=2 N-tiles per wave -> 4 independent WMMA chains per k-step.
__global__ void k_gemm_gated(const ushort_t* __restrict__ A, int strideA,
                             const ushort_t* __restrict__ BpG,
                             const ushort_t* __restrict__ BpW,
                             float* __restrict__ outF,
                             int M, int Np, int Kp) {
  const int lane = threadIdx.x & 31;
  const int wave = threadIdx.x >> 5;
  const int row0 = (blockIdx.x * 8 + wave) * 16;
  const int nt0  = blockIdx.y * 2;
  const int NT   = Np >> 4;
  const int KT   = Kp >> 5;
  v8f cg[2], cw[2];
#pragma unroll
  for (int j = 0; j < 2; ++j) {
    cg[j] = v8f{0.f,0.f,0.f,0.f,0.f,0.f,0.f,0.f};
    cw[j] = v8f{0.f,0.f,0.f,0.f,0.f,0.f,0.f,0.f};
  }
  for (int kt = 0; kt < KT; ++kt) {
    v16bf a = load_afrag(A, row0, M, strideA, kt << 5, lane);
#pragma unroll
    for (int j = 0; j < 2; ++j) {
      v16bf bg = load_bfrag(BpG, kt, nt0 + j, NT, lane);
      v16bf bw = load_bfrag(BpW, kt, nt0 + j, NT, lane);
      cg[j] = __builtin_amdgcn_wmma_f32_16x16x32_bf16(false, a, false, bg,
                                                      (short)0, cg[j], false, false);
      cw[j] = __builtin_amdgcn_wmma_f32_16x16x32_bf16(false, a, false, bw,
                                                      (short)0, cw[j], false, false);
    }
  }
  const int mb = row0 + ((lane >> 4) << 3);
#pragma unroll
  for (int j = 0; j < 2; ++j) {
    const int n = ((nt0 + j) << 4) + (lane & 15);
#pragma unroll
    for (int i = 0; i < 8; ++i) {
      int m = mb + i;
      if (m < M) {
        float g = 1.f / (1.f + __expf(-cg[j][i]));
        outF[(size_t)m * Np + n] = g * cw[j][i];
      }
    }
  }
}

// ---------------------------------------------------------------------------
// Weight pack: f32 [rows of (rowOff..rowOff+K) x N, ld=ldw] -> WMMA B tiles.
// Per tile (kt,nt): lane L element e maps to (k = kt*32 + (L>>4)*8 + (e&7) + (e>>3)*16,
//                                             n = nt*16 + (L&15)).
// ---------------------------------------------------------------------------
__global__ void k_pack_w(const float* __restrict__ W, int ldw, int rowOff,
                         int K, int N, ushort_t* __restrict__ out,
                         int KT, int NT) {
  size_t idx   = (size_t)blockIdx.x * 256 + threadIdx.x;
  size_t total = (size_t)KT * NT * 512;
  if (idx >= total) return;
  int e     = (int)(idx & 15);
  int lane  = (int)((idx >> 4) & 31);
  size_t tl = idx >> 9;
  int nt = (int)(tl % NT);
  int kt = (int)(tl / NT);
  int k  = (kt << 5) + ((lane >> 4) << 3) + (e & 7) + ((e >> 3) << 4);
  int n  = (nt << 4) + (lane & 15);
  float v = (k < K && n < N) ? W[(size_t)(rowOff + k) * ldw + n] : 0.f;
  out[idx] = f2b(v);
}

// f32 [M,K] -> bf16 [M,KP], zero-padded K
__global__ void k_cvt_pad(const float* __restrict__ X, ushort_t* __restrict__ Y,
                          int M, int K, int KP) {
  size_t idx   = (size_t)blockIdx.x * 256 + threadIdx.x;
  size_t total = (size_t)M * KP;
  if (idx >= total) return;
  int kp   = (int)(idx % KP);
  size_t m = idx / KP;
  Y[idx] = (kp < K) ? f2b(X[m * K + kp]) : (ushort_t)0;
}

__global__ void k_relu_bf16(const float* __restrict__ X,
                            ushort_t* __restrict__ Y, size_t n) {
  size_t i = (size_t)blockIdx.x * 256 + threadIdx.x;
  if (i < n) Y[i] = f2b(fmaxf(X[i], 0.f));
}

// Edge scatter: acc[dst] += Msg[src]; 64 threads/edge, float4 each
__global__ void k_scatter(const float* __restrict__ Msg,
                          const int* __restrict__ src,
                          const int* __restrict__ dst,
                          float* __restrict__ acc, int nEdges) {
  int g = threadIdx.x >> 6;
  int t = threadIdx.x & 63;
  int e = blockIdx.x * 4 + g;
  if (e >= nEdges) return;
  int s = src[e], d = dst[e];
  const float4 v = ((const float4*)(Msg + (size_t)s * HID))[t];
  float* ad = acc + (size_t)d * HID + t * 4;
  atomicAdd(ad + 0, v.x);
  atomicAdd(ad + 1, v.y);
  atomicAdd(ad + 2, v.z);
  atomicAdd(ad + 3, v.w);
}

// dr[b, cOff+c] = sum_o s2w[o] * |Xf[b*128+o,c] - Xs[b*128+o,c]|
__global__ void k_absdiff_reduce(const ushort_t* __restrict__ Xf,
                                 const ushort_t* __restrict__ Xs,
                                 const float* __restrict__ s2w,
                                 float* __restrict__ dr, int cOff, int drLD) {
  int b = blockIdx.x, c = threadIdx.x;
  float acc = 0.f;
  for (int o = 0; o < NOBJ; ++o) {
    size_t n = (size_t)(b * NOBJ + o) * HID + c;
    acc += s2w[o] * fabsf(b2f(Xf[n]) - b2f(Xs[n]));
  }
  dr[(size_t)b * drLD + cOff + c] = acc;
}

// bias_final[f] = s1_b[f] * sum(s2w) + s2_b[0]
__global__ void k_bias_final(const float* __restrict__ s1b,
                             const float* __restrict__ s2w,
                             const float* __restrict__ s2b,
                             float* __restrict__ bf) {
  __shared__ float sw;
  if (threadIdx.x == 0) {
    float s = 0.f;
    for (int o = 0; o < NOBJ; ++o) s += s2w[o];
    sw = s;
  }
  __syncthreads();
  bf[threadIdx.x] = s1b[threadIdx.x] * sw + s2b[0];
}

__global__ void k_pad_bias(const float* __restrict__ b, int n,
                           float* __restrict__ out, int np) {
  int i = threadIdx.x;
  if (i < np) out[i] = (i < n) ? b[i] : 0.f;
}

// softmax over first 10 of logits[256,16]; writes action + one_hot (padded 16)
__global__ void k_softmax_onehot(const float* __restrict__ logits,
                                 float* __restrict__ action,
                                 float* __restrict__ onehot) {
  int b = threadIdx.x;
  const float* l = logits + b * 16;
  float mx = l[0]; int am = 0;
  for (int i = 1; i < 10; ++i) if (l[i] > mx) { mx = l[i]; am = i; }
  float e[10], s = 0.f;
  for (int i = 0; i < 10; ++i) { e[i] = __expf(l[i] - mx); s += e[i]; }
  for (int i = 0; i < 10; ++i) action[b * 10 + i] = e[i] / s;
  for (int i = 0; i < 16; ++i) onehot[b * 16 + i] = (i == am) ? 1.f : 0.f;
}

// pred_in[b, 0:1024] = bf16(final[b, :])
__global__ void k_predin1(const float* __restrict__ finalF,
                          ushort_t* __restrict__ predin) {
  size_t idx = (size_t)blockIdx.x * 256 + threadIdx.x;  // 256*1024
  int c = (int)(idx & 1023);
  int b = (int)(idx >> 10);
  predin[(size_t)b * 1056 + c] = f2b(finalF[idx]);
}
// pred_in[b, 1024:1056] = one_hot (10) then zeros
__global__ void k_predin2(const float* __restrict__ onehot,
                          ushort_t* __restrict__ predin) {
  int idx = blockIdx.x * 256 + threadIdx.x;             // 256*32
  int c = idx & 31, b = idx >> 5;
  float v = (c < 16) ? onehot[b * 16 + c] : 0.f;        // [10..15] already 0
  predin[(size_t)b * 1056 + 1024 + c] = f2b(v);
}

// x[b*128+o, n] = prelu(U[b,n] + V[o,n] + bias[n] (+ extra[bo]*wlast[n]))
__global__ void k_bcast_add_prelu(const float* __restrict__ U,
                                  const float* __restrict__ V,
                                  const float* __restrict__ bias,
                                  const float* __restrict__ alphap,
                                  const float* __restrict__ extra,
                                  const float* __restrict__ wlast,
                                  ushort_t* __restrict__ out) {
  int row = blockIdx.x;
  int n   = threadIdx.x;
  int b = row >> 7, o = row & 127;
  float v = U[(size_t)b * HID + n] + V[(size_t)o * HID + n] + bias[n];
  if (extra) v += extra[row] * wlast[n];
  float a = *alphap;
  v = (v >= 0.f) ? v : a * v;
  out[(size_t)row * HID + n] = f2b(v);
}

// out[row] = sigmoid(dot(X[row,:256], w) + bias[0]); one wave per row
__global__ void k_rowdot_sigmoid(const ushort_t* __restrict__ X,
                                 const float* __restrict__ w,
                                 const float* __restrict__ bias,
                                 float* __restrict__ out, int M) {
  int wv = threadIdx.x >> 5, lane = threadIdx.x & 31;
  int row = blockIdx.x * 8 + wv;
  if (row >= M) return;
  const ushort_t* x = X + (size_t)row * HID;
  float acc = 0.f;
  for (int i = lane; i < HID; i += 32) acc += b2f(x[i]) * w[i];
  for (int off = 16; off; off >>= 1) acc += __shfl_down(acc, off, 32);
  if (lane == 0) out[row] = 1.f / (1.f + __expf(-(acc + bias[0])));
}

// out[b, 0:298] = [action(10) | pred1(128) | pred2_obj(128) | state(32)]
__global__ void k_assemble(const float* __restrict__ action,
                           const float* __restrict__ pred1,
                           const float* __restrict__ pred2o,
                           const float* __restrict__ stateF,
                           float* __restrict__ out) {
  int b = blockIdx.x, j = threadIdx.x;
  if (j >= 298) return;
  float v;
  if (j < 10)        v = action[b * 10 + j];
  else if (j < 138)  v = pred1[b * NOBJ + (j - 10)];
  else if (j < 266)  v = pred2o[b * NOBJ + (j - 138)];
  else               v = stateF[b * 32 + (j - 266)];
  out[(size_t)b * 298 + j] = v;
}

// ---------------------------------------------------------------------------
// Host-side helpers
// ---------------------------------------------------------------------------
static inline int cdiv(int a, int b) { return (a + b - 1) / b; }

static void packW(hipStream_t s, const float* W, int ldw, int rowOff,
                  int K, int N, ushort_t* out) {
  int KT = cdiv(K, 32), NT = cdiv(N, 16);
  size_t total = (size_t)KT * NT * 512;
  k_pack_w<<<(unsigned)((total + 255) / 256), 256, 0, s>>>(W, ldw, rowOff, K, N,
                                                           out, KT, NT);
}

template<int ACT>
static void gemm_act(hipStream_t s, const ushort_t* A, int strideA,
                     const ushort_t* Bp, const float* bias, float* outF,
                     ushort_t* outB, int M, int Np, int Kp, const float* alphap) {
  dim3 blk(256);
  if ((Np & 63) == 0) {
    dim3 g(cdiv(M, 128), Np / 64);
    k_gemm<ACT, 4><<<g, blk, 0, s>>>(A, strideA, Bp, bias, outF, outB, M, Np, Kp, alphap);
  } else if ((Np & 31) == 0) {
    dim3 g(cdiv(M, 128), Np / 32);
    k_gemm<ACT, 2><<<g, blk, 0, s>>>(A, strideA, Bp, bias, outF, outB, M, Np, Kp, alphap);
  } else {
    dim3 g(cdiv(M, 128), Np / 16);
    k_gemm<ACT, 1><<<g, blk, 0, s>>>(A, strideA, Bp, bias, outF, outB, M, Np, Kp, alphap);
  }
}

static void gemm(hipStream_t s, int act, const ushort_t* A, int strideA,
                 const ushort_t* Bp, const float* bias, float* outF,
                 ushort_t* outB, int M, int Np, int Kp, const float* alphap) {
  switch (act) {
    case 0: gemm_act<0>(s, A, strideA, Bp, bias, outF, outB, M, Np, Kp, alphap); break;
    case 1: gemm_act<1>(s, A, strideA, Bp, bias, outF, outB, M, Np, Kp, alphap); break;
    case 2: gemm_act<2>(s, A, strideA, Bp, bias, outF, outB, M, Np, Kp, alphap); break;
    default: gemm_act<3>(s, A, strideA, Bp, bias, outF, outB, M, Np, Kp, alphap); break;
  }
}

static void gnn_layer(hipStream_t s, const ushort_t* hA, int strideA, int Kw, int Kp,
                      const float* Wrel, const float* Grel, const float* Wself,
                      const float* bvec, const int* src, const int* dst,
                      float* accF, float* mrF, ushort_t* outB,
                      ushort_t* pw0, ushort_t* pw1) {
  packW(s, Wself, HID, 0, Kw, HID, pw0);
  gemm(s, 0, hA, strideA, pw0, bvec, accF, nullptr, NB, HID, Kp, nullptr);
  for (int r = 0; r < NETYPES; ++r) {
    packW(s, Grel + (size_t)r * Kw * HID, HID, 0, Kw, HID, pw0);
    packW(s, Wrel + (size_t)r * Kw * HID, HID, 0, Kw, HID, pw1);
    k_gemm_gated<<<dim3(NB / 128, HID / 32), 256, 0, s>>>(hA, strideA, pw0, pw1,
                                                          mrF, NB, HID, Kp);
    k_scatter<<<NEDGES / 4, 256, 0, s>>>(mrF, src + (size_t)r * NEDGES,
                                         dst + (size_t)r * NEDGES, accF, NEDGES);
  }
  k_relu_bf16<<<(unsigned)(((size_t)NB * HID + 255) / 256), 256, 0, s>>>(
      accF, outB, (size_t)NB * HID);
}

// ---------------------------------------------------------------------------
// kernel_launch
// ---------------------------------------------------------------------------
extern "C" void kernel_launch(void* const* d_in, const int* in_sizes, int n_in,
                              void* d_out, int out_size, void* d_ws, size_t ws_size,
                              hipStream_t stream) {
  (void)in_sizes; (void)n_in; (void)out_size; (void)ws_size;
  hipStream_t s = stream;

  const float* feat_g    = (const float*)d_in[0];
  const float* feat_goal = (const float*)d_in[1];
  const int*   src_g     = (const int*)d_in[2];
  const int*   dst_g     = (const int*)d_in[3];
  const int*   src_goal  = (const int*)d_in[4];
  const int*   dst_goal  = (const int*)d_in[5];
  // params, flattened in dict-insertion order
  const float* e1l0_Wrel = (const float*)d_in[7];
  const float* e1l0_Grel = (const float*)d_in[8];
  const float* e1l0_Ws   = (const float*)d_in[9];
  const float* e1l0_b    = (const float*)d_in[10];
  const float* e1l1_Wrel = (const float*)d_in[11];
  const float* e1l1_Grel = (const float*)d_in[12];
  const float* e1l1_Ws   = (const float*)d_in[13];
  const float* e1l1_b    = (const float*)d_in[14];
  const float* e2l0_Wrel = (const float*)d_in[15];
  const float* e2l0_Grel = (const float*)d_in[16];
  const float* e2l0_Ws   = (const float*)d_in[17];
  const float* e2l0_b    = (const float*)d_in[18];
  const float* e2l1_Wrel = (const float*)d_in[19];
  const float* e2l1_Grel = (const float*)d_in[20];
  const float* e2l1_Ws   = (const float*)d_in[21];
  const float* e2l1_b    = (const float*)d_in[22];
  const float* m1w = (const float*)d_in[23]; const float* m1b = (const float*)d_in[24];
  const float* m2w = (const float*)d_in[25]; const float* m2b = (const float*)d_in[26];
  const float* e1w = (const float*)d_in[27]; const float* e1b = (const float*)d_in[28];
  const float* e2w = (const float*)d_in[29]; const float* e2b = (const float*)d_in[30];
  const float* s1w = (const float*)d_in[31]; const float* s1b = (const float*)d_in[32];
  const float* s2w = (const float*)d_in[33]; const float* s2b = (const float*)d_in[34];
  const float* fc1w = (const float*)d_in[35]; const float* fc1b = (const float*)d_in[36];
  const float* fc2w = (const float*)d_in[37]; const float* fc2b = (const float*)d_in[38];
  const float* fc3w = (const float*)d_in[39]; const float* fc3b = (const float*)d_in[40];
  const float* p1w = (const float*)d_in[41]; const float* p1b = (const float*)d_in[42];
  const float* p2w = (const float*)d_in[43]; const float* p2b = (const float*)d_in[44];
  const float* p3w = (const float*)d_in[45]; const float* p3b = (const float*)d_in[46];
  const float* q1ow = (const float*)d_in[47]; const float* q1ob = (const float*)d_in[48];
  const float* q2ow = (const float*)d_in[49]; const float* q2ob = (const float*)d_in[50];
  const float* q3ow = (const float*)d_in[51]; const float* q3ob = (const float*)d_in[52];
  const float* q1sw = (const float*)d_in[53]; const float* q1sb = (const float*)d_in[54];
  const float* q2sw = (const float*)d_in[55]; const float* q2sb = (const float*)d_in[56];
  const float* q3sw = (const float*)d_in[57]; const float* q3sb = (const float*)d_in[58];
  const float* alphap = (const float*)d_in[59];
  const float* objvec = (const float*)d_in[60];

  // --- workspace layout -----------------------------------------------------
  char* base = (char*)d_ws;
  size_t off = 0;
  auto alloc = [&](size_t bytes) -> char* {
    char* p = base + off;
    off = (off + bytes + 255) & ~((size_t)255);
    return p;
  };
  ushort_t* featB = (ushort_t*)alloc((size_t)NB * 320 * 2);
  ushort_t* h1B   = (ushort_t*)alloc((size_t)NB * HID * 2);
  float*    accF  = (float*)alloc((size_t)NB * HID * 4);
  float*    mrF   = (float*)alloc((size_t)NB * HID * 4);
  ushort_t* hg_s  = (ushort_t*)alloc((size_t)NB * HID * 2);
  ushort_t* m_s   = (ushort_t*)alloc((size_t)NB * HID * 2);
  ushort_t* hg_f  = (ushort_t*)alloc((size_t)NB * HID * 2);
  ushort_t* m_f   = (ushort_t*)alloc((size_t)NB * HID * 2);
  ushort_t* pw0   = (ushort_t*)alloc(2u << 20);
  ushort_t* pw1   = (ushort_t*)alloc(2u << 20);
  float*    drF     = (float*)alloc(256 * 512 * 4);
  ushort_t* drB     = (ushort_t*)alloc(256 * 512 * 2);
  float*    biasF   = (float*)alloc(1024 * 4);
  float*    finalF  = (float*)alloc(256 * 1024 * 4);
  ushort_t* predin  = (ushort_t*)alloc(256 * 1056 * 2);
  ushort_t* xf1     = (ushort_t*)alloc(256 * 256 * 2);
  ushort_t* xf2     = (ushort_t*)alloc(256 * 256 * 2);
  float*    logits  = (float*)alloc(256 * 16 * 4);
  float*    actionB = (float*)alloc(256 * 10 * 4);
  float*    onehot  = (float*)alloc(256 * 16 * 4);
  float*    biasPad = (float*)alloc(16 * 4);
  ushort_t* objFtB  = (ushort_t*)alloc(128 * 320 * 2);
  ushort_t* obj1B   = (ushort_t*)alloc(128 * 256 * 2);
  ushort_t* objB    = (ushort_t*)alloc(128 * 256 * 2);
  float*    U1      = (float*)alloc(256 * 256 * 4);
  float*    V1      = (float*)alloc(128 * 256 * 4);
  float*    U2      = (float*)alloc(256 * 256 * 4);
  float*    V2      = (float*)alloc(128 * 256 * 4);
  float*    pred1   = (float*)alloc((size_t)NB * 4);
  float*    pred2o  = (float*)alloc((size_t)NB * 4);
  ushort_t* sB      = (ushort_t*)alloc(256 * 256 * 2);
  ushort_t* s2B     = (ushort_t*)alloc(256 * 256 * 2);
  float*    stateF  = (float*)alloc(256 * 32 * 4);
  // per-object head buffers reuse the freed GNN accumulators (stream-ordered)
  ushort_t* xA = (ushort_t*)accF;
  ushort_t* xB = (ushort_t*)mrF;

  const unsigned cvtFeatBlocks = (unsigned)(((size_t)NB * 320 + 255) / 256);

  // ===================== Encoder 1 (feat_g) =====================
  k_cvt_pad<<<cvtFeatBlocks, 256, 0, s>>>(feat_g, featB, NB, 300, 320);
  gnn_layer(s, featB, 320, 300, 320, e1l0_Wrel, e1l0_Grel, e1l0_Ws, e1l0_b,
            src_g, dst_g, accF, mrF, h1B, pw0, pw1);
  gnn_layer(s, h1B, 256, 256, 256, e1l1_Wrel, e1l1_Grel, e1l1_Ws, e1l1_b,
            src_g, dst_g, accF, mrF, hg_s, pw0, pw1);
  packW(s, m1w, HID, 0, 300, HID, pw0);
  gemm(s, 2, featB, 320, pw0, m1b, nullptr, h1B, NB, HID, 320, alphap);
  packW(s, m2w, HID, 0, 256, HID, pw0);
  gemm(s, 2, h1B, 256, pw0, m2b, nullptr, m_s, NB, HID, 256, alphap);

  // ===================== Encoder 2 (feat_goal) =====================
  k_cvt_pad<<<cvtFeatBlocks, 256, 0, s>>>(feat_goal, featB, NB, 300, 320);
  gnn_layer(s, featB, 320, 300, 320, e2l0_Wrel, e2l0_Grel, e2l0_Ws, e2l0_b,
            src_goal, dst_goal, accF, mrF, h1B, pw0, pw1);
  gnn_layer(s, h1B, 256, 256, 256, e2l1_Wrel, e2l1_Grel, e2l1_Ws, e2l1_b,
            src_goal, dst_goal, accF, mrF, hg_f, pw0, pw1);
  packW(s, m1w, HID, 0, 300, HID, pw0);
  gemm(s, 2, featB, 320, pw0, m1b, nullptr, h1B, NB, HID, 320, alphap);
  packW(s, m2w, HID, 0, 256, HID, pw0);
  gemm(s, 2, h1B, 256, pw0, m2b, nullptr, m_f, NB, HID, 256, alphap);

  // ===================== Scene: object-reduced |diff| then s1 GEMM ==========
  k_absdiff_reduce<<<256, 256, 0, s>>>(hg_f, hg_s, s2w, drF, 0, 512);
  k_absdiff_reduce<<<256, 256, 0, s>>>(m_f, m_s, s2w, drF, 256, 512);
  k_bias_final<<<1, 1024, 0, s>>>(s1b, s2w, s2b, biasF);
  k_cvt_pad<<<(256 * 512 + 255) / 256, 256, 0, s>>>(drF, drB, 256, 512, 512);
  packW(s, s1w, 1024, 0, 512, 1024, pw0);
  gemm(s, 0, drB, 512, pw0, biasF, finalF, nullptr, 256, 1024, 512, nullptr);
  k_predin1<<<(256 * 1024) / 256, 256, 0, s>>>(finalF, predin);

  // ===================== Action head =====================
  packW(s, fc1w, HID, 0, 1024, HID, pw0);
  gemm(s, 2, predin, 1056, pw0, fc1b, nullptr, xf1, 256, HID, 1024, alphap);
  packW(s, fc2w, HID, 0, 256, HID, pw0);
  gemm(s, 2, xf1, 256, pw0, fc2b, nullptr, xf2, 256, HID, 256, alphap);
  k_pad_bias<<<1, 32, 0, s>>>(fc3b, 10, biasPad, 16);
  packW(s, fc3w, 10, 0, 256, 10, pw0);
  gemm(s, 0, xf2, 256, pw0, biasPad, logits, nullptr, 256, 16, 256, nullptr);
  k_softmax_onehot<<<1, 256, 0, s>>>(logits, actionB, onehot);
  k_predin2<<<32, 256, 0, s>>>(onehot, predin);

  // ===================== Object embedding =====================
  k_cvt_pad<<<(128 * 320 + 255) / 256, 256, 0, s>>>(objvec, objFtB, 128, 300, 320);
  packW(s, e1w, HID, 0, 300, HID, pw0);
  gemm(s, 2, objFtB, 320, pw0, e1b, nullptr, obj1B, 128, HID, 320, alphap);
  packW(s, e2w, HID, 0, 256, HID, pw0);
  gemm(s, 2, obj1B, 256, pw0, e2b, nullptr, objB, 128, HID, 256, alphap);

  // ===================== pred1 head (factored concat) =====================
  packW(s, p1w, HID, 0, 1034, HID, pw0);            // rows [0,1034)
  gemm(s, 0, predin, 1056, pw0, nullptr, U1, nullptr, 256, HID, 1056, nullptr);
  packW(s, p1w, HID, 1034, 256, HID, pw1);          // rows [1034,1290)
  gemm(s, 0, objB, 256, pw1, nullptr, V1, nullptr, 128, HID, 256, nullptr);
  k_bcast_add_prelu<<<NB, 256, 0, s>>>(U1, V1, p1b, alphap, nullptr, nullptr, xA);
  packW(s, p2w, HID, 0, 256, HID, pw0);
  gemm(s, 2, xA, 256, pw0, p2b, nullptr, xB, NB, HID, 256, alphap);
  k_rowdot_sigmoid<<<NB / 8, 256, 0, s>>>(xB, p3w, p3b, pred1, NB);

  // ===================== pred2 object head =====================
  packW(s, q1ow, HID, 0, 1034, HID, pw0);
  gemm(s, 0, predin, 1056, pw0, nullptr, U2, nullptr, 256, HID, 1056, nullptr);
  packW(s, q1ow, HID, 1034, 256, HID, pw1);
  gemm(s, 0, objB, 256, pw1, nullptr, V2, nullptr, 128, HID, 256, nullptr);
  k_bcast_add_prelu<<<NB, 256, 0, s>>>(U2, V2, q1ob, alphap, pred1,
                                       q1ow + (size_t)1290 * HID, xA);
  packW(s, q2ow, HID, 0, 256, HID, pw0);
  gemm(s, 2, xA, 256, pw0, q2ob, nullptr, xB, NB, HID, 256, alphap);
  k_rowdot_sigmoid<<<NB / 8, 256, 0, s>>>(xB, q3ow, q3ob, pred2o, NB);

  // ===================== pred2 state head =====================
  packW(s, q1sw, HID, 0, 1034, HID, pw0);
  gemm(s, 2, predin, 1056, pw0, q1sb, nullptr, sB, 256, HID, 1056, alphap);
  packW(s, q2sw, HID, 0, 256, HID, pw0);
  gemm(s, 2, sB, 256, pw0, q2sb, nullptr, s2B, 256, HID, 256, alphap);
  packW(s, q3sw, 32, 0, 256, 32, pw0);
  gemm(s, 3, s2B, 256, pw0, q3sb, stateF, nullptr, 256, 32, 256, nullptr);

  // ===================== assemble output [256, 298] =====================
  k_assemble<<<256, 320, 0, s>>>(actionB, pred1, pred2o, stateF, (float*)d_out);
}
